// MultiheadAttention_32358283608545
// MI455X (gfx1250) — matrix operand order repaired and checked by reading, not compiled, not run on hardware
//
#include <hip/hip_runtime.h>
#include <hip/hip_bf16.h>

typedef _Float16 h16_t;
typedef __attribute__((ext_vector_type(16))) _Float16 v16h;
typedef __attribute__((ext_vector_type(8)))  _Float16 v8h;
typedef __attribute__((ext_vector_type(8)))  float    v8f;
typedef __attribute__((ext_vector_type(4)))  float    v4f;

#define WMMA_F16(a, b, c) \
  __builtin_amdgcn_wmma_f32_16x16x32_f16(false, (a), false, (b), (short)0, (c), false, false)

// ---------------------------------------------------------------------------
// Fragment loaders (wave32, CDNA5 16x16x32 f16 WMMA layouts per ISA 7.12.2)
// ---------------------------------------------------------------------------

// A fragment (16x32 f16) from a row-major f16 matrix, tile origin (row0, k0).
__device__ __forceinline__ v16h load_a_f16(const h16_t* base, int ld, int row0, int k0) {
  const int lane = threadIdx.x & 31;
  const int m    = lane & 15;
  const int kb   = (lane < 16) ? 0 : 8;
  const h16_t* p = base + (size_t)(row0 + m) * ld + k0 + kb;
  v8h lo = *(const v8h*)p;
  v8h hi = *(const v8h*)(p + 16);
  v16h r;
#pragma unroll
  for (int i = 0; i < 8; ++i) { r[i] = lo[i]; r[i + 8] = hi[i]; }
  return r;
}

// A fragment from a row-major f32 matrix, converting to f16 on the fly.
__device__ __forceinline__ v16h load_a_f32(const float* base, int ld, int row0, int k0) {
  const int lane = threadIdx.x & 31;
  const int m    = lane & 15;
  const int kb   = (lane < 16) ? 0 : 8;
  const float* p = base + (size_t)(row0 + m) * ld + k0 + kb;
  v4f a0 = *(const v4f*)p;
  v4f a1 = *(const v4f*)(p + 4);
  v4f a2 = *(const v4f*)(p + 16);
  v4f a3 = *(const v4f*)(p + 20);
  v16h r;
#pragma unroll
  for (int i = 0; i < 4; ++i) {
    r[i]      = (h16_t)a0[i];
    r[i + 4]  = (h16_t)a1[i];
    r[i + 8]  = (h16_t)a2[i];
    r[i + 12] = (h16_t)a3[i];
  }
  return r;
}

// B fragment (32x16) where element (k, n) is stored at base[n*ld + k].
__device__ __forceinline__ v16h load_b_kmaj(const h16_t* base, int ld, int col0, int k0) {
  const int lane = threadIdx.x & 31;
  const int n    = lane & 15;
  const int kb   = (lane < 16) ? 0 : 8;
  const h16_t* p = base + (size_t)(col0 + n) * ld + k0 + kb;
  v8h lo = *(const v8h*)p;
  v8h hi = *(const v8h*)(p + 16);
  v16h r;
#pragma unroll
  for (int i = 0; i < 8; ++i) { r[i] = lo[i]; r[i + 8] = hi[i]; }
  return r;
}

// ---------------------------------------------------------------------------
// K0: transpose + convert a 512x512 f32 weight matrix to f16 [out][in]
// ---------------------------------------------------------------------------
__global__ __launch_bounds__(256) void wtrans_kernel(const float* __restrict__ w,
                                                     h16_t* __restrict__ wT) {
  int idx = blockIdx.x * 256 + threadIdx.x;
  int d = idx >> 9;
  int e = idx & 511;
  wT[(size_t)e * 512 + d] = (h16_t)w[idx];
}

// ---------------------------------------------------------------------------
// K1: projection GEMM  [B*N,512] x [512,512] -> f16, double-buffered pipeline
// ---------------------------------------------------------------------------
__global__ __launch_bounds__(256) void proj_kernel(const float* __restrict__ x,
                                                   const h16_t* __restrict__ wT,
                                                   const float* __restrict__ bias,
                                                   h16_t* __restrict__ out, int mode) {
  const int row0  = blockIdx.x * 16;
  const int wave  = threadIdx.x >> 5;
  const int lane  = threadIdx.x & 31;
  const int n     = lane & 15;
  const int mbase = (lane < 16) ? 0 : 8;
  const int e0b   = wave * 64;

  v8f  acc[4];
  v16h a_buf[2];
  v16h b_buf[2][4];
#pragma unroll
  for (int ct = 0; ct < 4; ++ct) acc[ct] = (v8f){};

  a_buf[0] = load_a_f32(x, 512, row0, 0);
#pragma unroll
  for (int ct = 0; ct < 4; ++ct) b_buf[0][ct] = load_b_kmaj(wT, 512, e0b + ct * 16, 0);

#pragma unroll
  for (int es = 0; es < 16; ++es) {
    const int cur = es & 1, nxt = cur ^ 1;
    if (es < 15) {
      a_buf[nxt] = load_a_f32(x, 512, row0, (es + 1) * 32);
#pragma unroll
      for (int ct = 0; ct < 4; ++ct)
        b_buf[nxt][ct] = load_b_kmaj(wT, 512, e0b + ct * 16, (es + 1) * 32);
    }
#pragma unroll
    for (int ct = 0; ct < 4; ++ct) acc[ct] = WMMA_F16(a_buf[cur], b_buf[cur][ct], acc[ct]);
  }

#pragma unroll
  for (int ct = 0; ct < 4; ++ct) {
    const int e0 = e0b + ct * 16;
    const float bv = bias[e0 + n];
    if (mode == 2) {
      const int b  = row0 >> 12;
      const int r0 = row0 & 4095;
#pragma unroll
      for (int j = 0; j < 8; ++j) {
        const int m = mbase + j;
        out[(size_t)b * (512u * 4096u) + (size_t)(e0 + n) * 4096 + (r0 + m)] =
            (h16_t)(acc[ct][j] + bv);
      }
    } else {
      const float scale = (mode == 0) ? 0.125f : 1.0f;
#pragma unroll
      for (int j = 0; j < 8; ++j) {
        const int m = mbase + j;
        out[(size_t)(row0 + m) * 512 + (e0 + n)] = (h16_t)((acc[ct][j] + bv) * scale);
      }
    }
  }
}

// ---------------------------------------------------------------------------
// K2: fused attention, 32-query block per workgroup (2 stacked 16-row tiles
// share every KP/VP B-fragment -> half the L2 traffic of a 16-row block).
//   LDS: S strip 32 x 4104 f16 (~256.5 KB) + staged QP 32x520 f16 (~32.5 KB)
// ---------------------------------------------------------------------------
#define NPAD 4104  // 4096 + 8 halfs: 16B-aligned rows, skewed banking
#define SQLD 520   // 512 + 8 halfs

__global__ __launch_bounds__(256) void attn_kernel(const h16_t* __restrict__ qp,
                                                   const h16_t* __restrict__ kp,
                                                   const h16_t* __restrict__ vpT,
                                                   h16_t* __restrict__ dpa) {
  __shared__ h16_t s_p[32 * NPAD];   // score/probability strip (2 query tiles)
  __shared__ h16_t s_q[32 * SQLD];   // staged QP block
  __shared__ float s_inv[32];

  const int b     = blockIdx.y;
  const int q0    = blockIdx.x * 32;
  const int wave  = threadIdx.x >> 5;
  const int lane  = threadIdx.x & 31;
  const int n     = lane & 15;
  const int mbase = (lane < 16) ? 0 : 8;

  const h16_t* qpb = qp + (size_t)b * 4096 * 512;
  const h16_t* kpb = kp + (size_t)b * 4096 * 512;
  const h16_t* vtb = vpT + (size_t)b * 512 * 4096;

  // ---- Phase 0: stage QP[q0:32][512] into LDS (v8h copies) ----
  {
    const int t = threadIdx.x;
#pragma unroll
    for (int i = 0; i < 8; ++i) {
      const int idx = t + i * 256;       // 0..2047
      const int row = idx >> 6;
      const int col = idx & 63;
      ((v8h*)(s_q + row * SQLD))[col] =
          ((const v8h*)(qpb + (size_t)(q0 + row) * 512))[col];
    }
  }
  __syncthreads();

  // ---- Phase 1: S strip into LDS; wave covers keys [wave*512, +512).
  //      Both query tiles computed per key tile, sharing each B fragment. ----
  for (int ct = 0; ct < 32; ++ct) {
    const int key0 = wave * 512 + ct * 16;
    if (ct < 31)
      __builtin_prefetch(kpb + (size_t)(key0 + 16 + n) * 512, 0, 3);

    v16h b_buf[4];                    // depth-4 ring of KP fragments
#pragma unroll
    for (int i = 0; i < 4; ++i) b_buf[i] = load_b_kmaj(kpb, 512, key0, i * 32);
    v16h a0_buf[2], a1_buf[2];        // double-buffered A from LDS
    a0_buf[0] = load_a_f16(s_q, SQLD, 0, 0);
    a1_buf[0] = load_a_f16(s_q, SQLD, 16, 0);

    v8f acc0 = {}, acc1 = {};
#pragma unroll
    for (int es = 0; es < 16; ++es) {
      const int cur = es & 1, nxt = cur ^ 1;
      if (es < 15) {
        a0_buf[nxt] = load_a_f16(s_q, SQLD, 0, (es + 1) * 32);
        a1_buf[nxt] = load_a_f16(s_q, SQLD, 16, (es + 1) * 32);
      }
      acc0 = WMMA_F16(a0_buf[cur], b_buf[es & 3], acc0);
      acc1 = WMMA_F16(a1_buf[cur], b_buf[es & 3], acc1);
      if (es < 12) b_buf[es & 3] = load_b_kmaj(kpb, 512, key0, (es + 4) * 32);
    }
#pragma unroll
    for (int j = 0; j < 8; ++j) {
      const int m = mbase + j;
      s_p[m * NPAD + key0 + n]        = (h16_t)acc0[j];
      s_p[(16 + m) * NPAD + key0 + n] = (h16_t)acc1[j];
    }
  }
  __syncthreads();

  // ---- Phase 2: row softmax, vectorized v8h row scans (4 rows per wave) ----
#pragma unroll
  for (int i = 0; i < 4; ++i) {
    const int r = wave * 4 + i;
    v8h* rowv = (v8h*)(s_p + r * NPAD);
    float mx = -3.0e38f;
    for (int it = 0; it < 16; ++it) {
      v8h xv = rowv[it * 32 + lane];
#pragma unroll
      for (int j = 0; j < 8; ++j) mx = fmaxf(mx, (float)xv[j]);
    }
#pragma unroll
    for (int off = 16; off > 0; off >>= 1) mx = fmaxf(mx, __shfl_xor(mx, off, 32));

    float sum = 0.0f;
    for (int it = 0; it < 16; ++it) {
      v8h xv = rowv[it * 32 + lane];
      v8h pv;
#pragma unroll
      for (int j = 0; j < 8; ++j) {
        float p = __expf((float)xv[j] - mx);
        pv[j] = (h16_t)p;
        sum += p;
      }
      rowv[it * 32 + lane] = pv;
    }
#pragma unroll
    for (int off = 16; off > 0; off >>= 1) sum += __shfl_xor(sum, off, 32);
    if (lane == 0) s_inv[r] = 1.0f / sum;
  }
  __syncthreads();

  // ---- Phase 3: O = P @ VP; wave covers value cols [wave*64, +64),
  //      both query tiles share each VP B-fragment. ----
  v8f  oacc[2][4];
  v16h a_buf[2][2];                   // [query tile][ring]
  v16h b_buf[2][4];
#pragma unroll
  for (int qt = 0; qt < 2; ++qt)
#pragma unroll
    for (int ct = 0; ct < 4; ++ct) oacc[qt][ct] = (v8f){};

  a_buf[0][0] = load_a_f16(s_p, NPAD, 0, 0);
  a_buf[1][0] = load_a_f16(s_p, NPAD, 16, 0);
#pragma unroll
  for (int ct = 0; ct < 4; ++ct) b_buf[0][ct] = load_b_kmaj(vtb, 4096, wave * 64 + ct * 16, 0);

#pragma unroll 2
  for (int kt = 0; kt < 128; ++kt) {
    const int cur = kt & 1, nxt = cur ^ 1;
    if (kt < 127) {
      a_buf[0][nxt] = load_a_f16(s_p, NPAD, 0, (kt + 1) * 32);
      a_buf[1][nxt] = load_a_f16(s_p, NPAD, 16, (kt + 1) * 32);
#pragma unroll
      for (int ct = 0; ct < 4; ++ct)
        b_buf[nxt][ct] = load_b_kmaj(vtb, 4096, wave * 64 + ct * 16, (kt + 1) * 32);
    }
#pragma unroll
    for (int ct = 0; ct < 4; ++ct) {
      oacc[0][ct] = WMMA_F16(a_buf[0][cur], b_buf[cur][ct], oacc[0][ct]);
      oacc[1][ct] = WMMA_F16(a_buf[1][cur], b_buf[cur][ct], oacc[1][ct]);
    }
  }

#pragma unroll
  for (int qt = 0; qt < 2; ++qt)
#pragma unroll
    for (int ct = 0; ct < 4; ++ct) {
      const int e0 = wave * 64 + ct * 16;
#pragma unroll
      for (int j = 0; j < 8; ++j) {
        const int m = mbase + j;
        const int r = qt * 16 + m;
        dpa[(size_t)b * 4096 * 512 + (size_t)(q0 + r) * 512 + e0 + n] =
            (h16_t)(oacc[qt][ct][j] * s_inv[r]);
      }
    }
}

// ---------------------------------------------------------------------------
// K3: output projection  OUT = DPA @ W_mha + b_mha  (f32 output)
// ---------------------------------------------------------------------------
__global__ __launch_bounds__(256) void outproj_kernel(const h16_t* __restrict__ dpa,
                                                      const h16_t* __restrict__ wT,
                                                      const float* __restrict__ bias,
                                                      float* __restrict__ out) {
  const int row0  = blockIdx.x * 16;
  const int wave  = threadIdx.x >> 5;
  const int lane  = threadIdx.x & 31;
  const int n     = lane & 15;
  const int mbase = (lane < 16) ? 0 : 8;
  const int d0b   = wave * 64;

  v8f  acc[4];
  v16h a_buf[2];
  v16h b_buf[2][4];
#pragma unroll
  for (int ct = 0; ct < 4; ++ct) acc[ct] = (v8f){};

  a_buf[0] = load_a_f16(dpa, 512, row0, 0);
#pragma unroll
  for (int ct = 0; ct < 4; ++ct) b_buf[0][ct] = load_b_kmaj(wT, 512, d0b + ct * 16, 0);

#pragma unroll
  for (int es = 0; es < 16; ++es) {
    const int cur = es & 1, nxt = cur ^ 1;
    if (es < 15) {
      a_buf[nxt] = load_a_f16(dpa, 512, row0, (es + 1) * 32);
#pragma unroll
      for (int ct = 0; ct < 4; ++ct)
        b_buf[nxt][ct] = load_b_kmaj(wT, 512, d0b + ct * 16, (es + 1) * 32);
    }
#pragma unroll
    for (int ct = 0; ct < 4; ++ct) acc[ct] = WMMA_F16(a_buf[cur], b_buf[cur][ct], acc[ct]);
  }

#pragma unroll
  for (int ct = 0; ct < 4; ++ct) {
    const int d0 = d0b + ct * 16;
    const float bv = bias[d0 + n];
#pragma unroll
    for (int j = 0; j < 8; ++j) {
      const int m = mbase + j;
      out[(size_t)(row0 + m) * 512 + d0 + n] = acc[ct][j] + bv;
    }
  }
}

// ---------------------------------------------------------------------------
extern "C" void kernel_launch(void* const* d_in, const int* in_sizes, int n_in,
                              void* d_out, int out_size, void* d_ws, size_t ws_size,
                              hipStream_t stream) {
  (void)in_sizes; (void)n_in; (void)out_size; (void)ws_size;
  const float* q     = (const float*)d_in[0];
  const float* k     = (const float*)d_in[1];
  const float* v     = (const float*)d_in[2];
  const float* w_q   = (const float*)d_in[3];
  const float* w_k   = (const float*)d_in[4];
  const float* w_v   = (const float*)d_in[5];
  const float* w_mha = (const float*)d_in[6];
  const float* b_q   = (const float*)d_in[7];
  const float* b_k   = (const float*)d_in[8];
  const float* b_v   = (const float*)d_in[9];
  const float* b_mha = (const float*)d_in[10];

  const size_t BN     = 4u * 4096u;
  const size_t BIG    = BN * 512u * 2u;        // f16 [B*N,512] = 16 MB
  const size_t WBYTES = 512u * 512u * 2u;

  char*  ws  = (char*)d_ws;
  size_t off = 0;
  auto alloc = [&](size_t bytes) {
    void* p = ws + off;
    off = (off + bytes + 255) & ~(size_t)255;
    return p;
  };
  h16_t* qp  = (h16_t*)alloc(BIG);
  h16_t* kp  = (h16_t*)alloc(BIG);
  h16_t* vpT = (h16_t*)alloc(BIG);
  h16_t* dpa = (h16_t*)alloc(BIG);
  h16_t* wqT = (h16_t*)alloc(WBYTES);
  h16_t* wkT = (h16_t*)alloc(WBYTES);
  h16_t* wvT = (h16_t*)alloc(WBYTES);
  h16_t* wmT = (h16_t*)alloc(WBYTES);

  wtrans_kernel<<<dim3(1024), 256, 0, stream>>>(w_q, wqT);
  wtrans_kernel<<<dim3(1024), 256, 0, stream>>>(w_k, wkT);
  wtrans_kernel<<<dim3(1024), 256, 0, stream>>>(w_v, wvT);
  wtrans_kernel<<<dim3(1024), 256, 0, stream>>>(w_mha, wmT);

  proj_kernel<<<dim3(1024), 256, 0, stream>>>(q, wqT, b_q, qp, 0);
  proj_kernel<<<dim3(1024), 256, 0, stream>>>(k, wkT, b_k, kp, 1);
  proj_kernel<<<dim3(1024), 256, 0, stream>>>(v, wvT, b_v, vpT, 2);

  // 32-query blocks: grid 128 x 4
  attn_kernel<<<dim3(128, 4), 256, 0, stream>>>(qp, kp, vpT, dpa);

  outproj_kernel<<<dim3(1024), 256, 0, stream>>>(dpa, wmT, b_mha, (float*)d_out);
}
